// FckModel_23897198035121
// MI455X (gfx1250) — matrix operand order repaired and checked by reading, not compiled, not run on hardware
//
#include <hip/hip_runtime.h>

// ---------------------------------------------------------------------------
// Scene-graph message passing on MI455X (gfx1250, wave32).
// Heavy GEMMs: v_wmma_f32_16x16x32_bf16 fed from double-buffered LDS tiles
// filled with global_load_async_to_lds_b128 (ASYNCcnt / s_wait_asynccnt).
// Steady-state K-loop is branch-free (last iteration peeled).
// ---------------------------------------------------------------------------

typedef __bf16 bf16;
typedef __attribute__((ext_vector_type(16))) __bf16 v16bf;
typedef __attribute__((ext_vector_type(8)))  __bf16 v8bf;
typedef __attribute__((ext_vector_type(8)))  float  v8f;

#define OBJ   2048
#define FEATS 2432
#define NBOX  4096
#define NEDGE 32768

// LDS tile layout (per 256-thread block): A 64 rows x 32 K, B 256 cols x 32 K.
// Row strides padded to 80B (5 x 16B): lane address stride 20 DWORDs mod 64
// banks has gcd 4 -> 16 distinct banks across a 16-lane group (conflict-free).
#define LDS_A_STRIDE 80
#define LDS_B_STRIDE 80
#define LDS_A_BYTES  (64 * LDS_A_STRIDE)               // 5120
#define LDS_B_BYTES  (256 * LDS_B_STRIDE)              // 20480
#define LDS_BUF      (LDS_A_BYTES + LDS_B_BYTES)       // 25600
#define LDS_TOTAL    (2 * LDS_BUF)                     // 51200 (<320KB WGP LDS)

// ---------------------------------------------------------------------------
// CDNA5 async copy: 16 bytes per lane, global -> LDS, tracked by ASYNCcnt.
// ---------------------------------------------------------------------------
static __device__ __forceinline__ void async_copy16(unsigned lds_off, const bf16* gptr) {
    asm volatile("global_load_async_to_lds_b128 %0, %1, off"
                 :: "v"(lds_off), "v"(gptr) : "memory");
}
static __device__ __forceinline__ void wait_async_le5() {
    asm volatile("s_wait_asynccnt 0x5" ::: "memory");
}
static __device__ __forceinline__ void wait_async_0() {
    asm volatile("s_wait_asynccnt 0x0" ::: "memory");
}

// Build a v16bf fragment from two 16B-aligned LDS halves (+0, +step bytes).
static __device__ __forceinline__ v16bf frag16(const char* p, int step) {
    v8bf lo = *(const v8bf*)(p);
    v8bf hi = *(const v8bf*)(p + step);
    return __builtin_shufflevector(lo, hi, 0,1,2,3,4,5,6,7,8,9,10,11,12,13,14,15);
}

// ---------------------------------------------------------------------------
// Shared WMMA main loop.
//   Block tile: 64 rows x 256 cols.  Wave w: rows (w>>2)*32..+32, cols (w&3)*64..+64.
//   acc[t][nt]: t = 16-row subtile (0/1), nt = 16-col subtile (0..3).
//   A layout per lane (16x32 bf16): lanes 0-15 M=0..15 K{0..7,16..23},
//                                   lanes 16-31 M=0..15 K{8..15,24..31}.
//   B layout per lane (32x16 bf16): lane n = col n K{0..7,16..23}, lane n+16 K{8..15,24..31}.
// ---------------------------------------------------------------------------
static __device__ __forceinline__ void gemm_mainloop(
        const bf16* __restrict__ A, const bf16* __restrict__ WT,
        const int* __restrict__ gather, int rowBase0, int colBase0, int K,
        char* shMem, v8f acc[2][4]) {
    const int t     = threadIdx.x;
    const int lane  = t & 31;
    const int wave  = t >> 5;
    const int cn    = lane & 15;
    const int khalf = lane >> 4;
    const int wRow  = (wave >> 2) * 32;          // wave's row offset in block tile
    const int wCol  = (wave & 3) * 64;           // wave's col offset in block tile
    const unsigned shBase = (unsigned)(size_t)shMem;

    // --- per-thread async-fill assignments --------------------------------
    // A tile: 64 rows x 64B; thread -> row t>>2, 16B segment t&3.
    const int ra = t >> 2, sa = t & 3;
    const size_t aRow = gather ? (size_t)gather[rowBase0 + ra]
                               : (size_t)(rowBase0 + ra);
    const bf16* aSrc = A + aRow * K + sa * 8;
    const unsigned aDst = shBase + ra * LDS_A_STRIDE + sa * 16;

    // B tile: 256 cols x 64B = 1024 chunks; thread -> chunks t + 256*j.
    const bf16* bSrc[4];
    unsigned    bDst[4];
    #pragma unroll
    for (int j = 0; j < 4; ++j) {
        const int c = t + 256 * j;
        const int col = c >> 2, sb = c & 3;
        bSrc[j] = WT + (size_t)(colBase0 + col) * K + sb * 8;
        bDst[j] = shBase + LDS_A_BYTES + col * LDS_B_STRIDE + sb * 16;
    }

    // Sources advance 32 K-elements (64B) per fill; no k0-dependent math.
    auto fill = [&](int buf) {
        const unsigned bo = buf * LDS_BUF;
        async_copy16(aDst + bo, aSrc);
        #pragma unroll
        for (int j = 0; j < 4; ++j) async_copy16(bDst[j] + bo, bSrc[j]);
        aSrc += 32;
        #pragma unroll
        for (int j = 0; j < 4; ++j) bSrc[j] += 32;
    };

    auto compute = [&](int buf) {
        const char* base = shMem + buf * LDS_BUF;
        const char* ap = base + (wRow + cn) * LDS_A_STRIDE + khalf * 16;
        v16bf a0 = frag16(ap, 32);
        v16bf a1 = frag16(ap + 16 * LDS_A_STRIDE, 32);
        #pragma unroll
        for (int nt = 0; nt < 4; ++nt) {
            const char* bp = base + LDS_A_BYTES
                           + (wCol + nt * 16 + cn) * LDS_B_STRIDE + khalf * 16;
            v16bf b = frag16(bp, 32);
            acc[0][nt] = __builtin_amdgcn_wmma_f32_16x16x32_bf16(
                false, a0, false, b, (short)0, acc[0][nt], false, false);
            acc[1][nt] = __builtin_amdgcn_wmma_f32_16x16x32_bf16(
                false, a1, false, b, (short)0, acc[1][nt], false, false);
        }
    };

    const int nIter = K >> 5;
    fill(0);

    // steady state: unconditional prefetch of the next K-slab
    for (int iter = 0; iter < nIter - 1; ++iter) {
        const int cur = iter & 1;
        __syncthreads();            // readers of buf[1-cur] finished
        fill(1 - cur);              // async prefetch next slab
        wait_async_le5();           // own cur-buffer writes landed
        __syncthreads();            // all waves' cur-buffer writes visible
        compute(cur);
    }

    // peeled last iteration: nothing left to prefetch
    __syncthreads();
    wait_async_0();
    __syncthreads();
    compute((nIter - 1) & 1);
}

// ---------------------------------------------------------------------------
// Elementwise f32 -> bf16 convert
// ---------------------------------------------------------------------------
__global__ void cvt_bf16_kernel(const float* __restrict__ in, bf16* __restrict__ out,
                                size_t n) {
    size_t i = (size_t)blockIdx.x * blockDim.x + threadIdx.x;
    size_t stride = (size_t)gridDim.x * blockDim.x;
    for (; i < n; i += stride) out[i] = (bf16)in[i];
}

// Transpose f32 [K,Nn] -> bf16 [Nn,K] via LDS tile (bank-conflict padded)
__global__ void transpose_bf16_kernel(const float* __restrict__ in, bf16* __restrict__ out,
                                      int K, int Nn) {
    __shared__ float tile[32][33];
    const int kb = blockIdx.y * 32, nb = blockIdx.x * 32;
    const int tx = threadIdx.x, ty = threadIdx.y;   // 32 x 8
    #pragma unroll
    for (int r = ty; r < 32; r += 8)
        tile[r][tx] = in[(size_t)(kb + r) * Nn + nb + tx];
    __syncthreads();
    #pragma unroll
    for (int r = ty; r < 32; r += 8)
        out[(size_t)(nb + r) * K + kb + tx] = (bf16)tile[tx][r];
}

__global__ void fill_f32_kernel(float* __restrict__ p, float v, int n) {
    int i = blockIdx.x * blockDim.x + threadIdx.x;
    if (i < n) p[i] = v;
}

// ---------------------------------------------------------------------------
// GEMM: C = A @ W (+ bias).  Grid exactly covers (M/64) x (Nn/256).
// ---------------------------------------------------------------------------
__global__ void __launch_bounds__(256)
gemm_bf16_kernel(const bf16* __restrict__ A, const bf16* __restrict__ WT,
                 const float* __restrict__ bias, float* __restrict__ C,
                 int Nn, int K) {
    __shared__ __align__(16) char shMem[LDS_TOTAL];
    const int nB = Nn >> 8;
    const int rowBase0 = (blockIdx.x / nB) * 64;
    const int colBase0 = (blockIdx.x % nB) * 256;

    v8f acc[2][4] = {};
    gemm_mainloop(A, WT, nullptr, rowBase0, colBase0, K, shMem, acc);

    const int lane = threadIdx.x & 31, wave = threadIdx.x >> 5;
    const int cn = lane & 15, khalf = lane >> 4;
    const int wRow = (wave >> 2) * 32, wCol = (wave & 3) * 64;

    float bv[4];
    #pragma unroll
    for (int nt = 0; nt < 4; ++nt)
        bv[nt] = bias ? bias[colBase0 + wCol + nt * 16 + cn] : 0.0f;

    #pragma unroll
    for (int tsub = 0; tsub < 2; ++tsub)
        #pragma unroll
        for (int r = 0; r < 8; ++r) {
            size_t orow = (size_t)(rowBase0 + wRow + tsub * 16 + r + 8 * khalf) * Nn;
            #pragma unroll
            for (int nt = 0; nt < 4; ++nt)
                C[orow + colBase0 + wCol + nt * 16 + cn] = acc[tsub][nt][r] + bv[nt];
        }
}

// ---------------------------------------------------------------------------
// Attention GEMM (rows gathered through edge[]), fused logit epilogue:
//   h = relu(rel_att + box[edge] @ Wa_box + b_a1);  logits[e] += h . Wa2
// ---------------------------------------------------------------------------
__global__ void __launch_bounds__(256)
att_gemm_kernel(const bf16* __restrict__ boxBf, const bf16* __restrict__ WT,
                const int* __restrict__ edge, const float* __restrict__ relAtt,
                const float* __restrict__ b_a1, const float* __restrict__ Wa2,
                float* __restrict__ logits, int D) {
    __shared__ __align__(16) char shMem[LDS_TOTAL];
    const int nB = D >> 8;
    const int eBase0   = (blockIdx.x / nB) * 64;
    const int colBase0 = (blockIdx.x % nB) * 256;

    v8f acc[2][4] = {};
    gemm_mainloop(boxBf, WT, edge, eBase0, colBase0, D, shMem, acc);

    const int lane = threadIdx.x & 31, wave = threadIdx.x >> 5;
    const int cn = lane & 15, khalf = lane >> 4;
    const int wRow = (wave >> 2) * 32, wCol = (wave & 3) * 64;

    float ba[4], w2[4];
    #pragma unroll
    for (int nt = 0; nt < 4; ++nt) {
        const int col = colBase0 + wCol + nt * 16 + cn;
        ba[nt] = b_a1[col];
        w2[nt] = Wa2[col];
    }

    #pragma unroll
    for (int tsub = 0; tsub < 2; ++tsub)
        #pragma unroll
        for (int r = 0; r < 8; ++r) {
            const int erow = eBase0 + wRow + tsub * 16 + r + 8 * khalf;
            const float* ra = relAtt + (size_t)erow * D + colBase0 + wCol;
            float pr = 0.0f;
            #pragma unroll
            for (int nt = 0; nt < 4; ++nt) {
                float h = acc[tsub][nt][r] + ra[nt * 16 + cn] + ba[nt];
                h = fmaxf(h, 0.0f);
                pr += h * w2[nt];
            }
            // reduce over the 16 lanes of this half-wave (rows split by khalf)
            pr += __shfl_xor(pr, 1);
            pr += __shfl_xor(pr, 2);
            pr += __shfl_xor(pr, 4);
            pr += __shfl_xor(pr, 8);
            if (cn == 0) atomicAdd(&logits[erow], pr);
        }
}

// ---------------------------------------------------------------------------
// Final GEMM: out = relu(box @ W_box + b_box + agg_s + agg_o)
// ---------------------------------------------------------------------------
__global__ void __launch_bounds__(256)
final_gemm_kernel(const bf16* __restrict__ A, const bf16* __restrict__ WT,
                  const float* __restrict__ bias,
                  const float* __restrict__ aggS, const float* __restrict__ aggO,
                  float* __restrict__ out, int Nn, int K) {
    __shared__ __align__(16) char shMem[LDS_TOTAL];
    const int nB = Nn >> 8;
    const int rowBase0 = (blockIdx.x / nB) * 64;
    const int colBase0 = (blockIdx.x % nB) * 256;

    v8f acc[2][4] = {};
    gemm_mainloop(A, WT, nullptr, rowBase0, colBase0, K, shMem, acc);

    const int lane = threadIdx.x & 31, wave = threadIdx.x >> 5;
    const int cn = lane & 15, khalf = lane >> 4;
    const int wRow = (wave >> 2) * 32, wCol = (wave & 3) * 64;

    float bv[4];
    #pragma unroll
    for (int nt = 0; nt < 4; ++nt) bv[nt] = bias[colBase0 + wCol + nt * 16 + cn];

    #pragma unroll
    for (int tsub = 0; tsub < 2; ++tsub)
        #pragma unroll
        for (int r = 0; r < 8; ++r) {
            size_t orow = (size_t)(rowBase0 + wRow + tsub * 16 + r + 8 * khalf) * Nn;
            #pragma unroll
            for (int nt = 0; nt < 4; ++nt) {
                size_t idx = orow + colBase0 + wCol + nt * 16 + cn;
                float v = acc[tsub][nt][r] + bv[nt] + aggS[idx] + aggO[idx];
                out[idx] = fmaxf(v, 0.0f);
            }
        }
}

// ---------------------------------------------------------------------------
// Segment softmax pieces
// ---------------------------------------------------------------------------
static __device__ __forceinline__ void atomicMaxF(float* addr, float v) {
    if (v >= 0.0f) atomicMax((int*)addr, __float_as_int(v));
    else           atomicMin((unsigned int*)addr, __float_as_uint(v));
}

__global__ void seg_max_kernel(const float* __restrict__ logits, const int* __restrict__ seg,
                               float* __restrict__ segmax, int E) {
    int e = blockIdx.x * blockDim.x + threadIdx.x;
    if (e < E) atomicMaxF(&segmax[seg[e]], logits[e]);
}

__global__ void seg_exp_kernel(float* __restrict__ logits /* in: logit, out: exp */,
                               const int* __restrict__ seg,
                               const float* __restrict__ segmax,
                               float* __restrict__ denom, int E) {
    int e = blockIdx.x * blockDim.x + threadIdx.x;
    if (e < E) {
        float v = __expf(logits[e] - segmax[seg[e]]);
        logits[e] = v;
        atomicAdd(&denom[seg[e]], v);
    }
}

// agg[seg[e], :] += (expv[e]/denom[seg[e]]) * msg[e, :]; one block per edge
__global__ void scatter_kernel(const float* __restrict__ expv, const float* __restrict__ denom,
                               const int* __restrict__ seg, const float* __restrict__ msg,
                               float* __restrict__ agg) {
    const int e = blockIdx.x;
    const int node = seg[e];
    const float alpha = expv[e] / (denom[node] + 1e-9f);
    const float* m = msg + (size_t)e * OBJ;
    float* a = agg + (size_t)node * OBJ;
    for (int d = threadIdx.x; d < OBJ; d += blockDim.x)
        atomicAdd(&a[d], alpha * m[d]);
}

// ---------------------------------------------------------------------------
// Host orchestration
// ---------------------------------------------------------------------------
extern "C" void kernel_launch(void* const* d_in, const int* in_sizes, int n_in,
                              void* d_out, int out_size, void* d_ws, size_t ws_size,
                              hipStream_t stream) {
    (void)in_sizes; (void)n_in; (void)out_size; (void)ws_size;

    const float* box    = (const float*)d_in[0];
    const float* rel    = (const float*)d_in[1];
    const int*   esub   = (const int*)d_in[2];
    const int*   eobj   = (const int*)d_in[3];
    const float* W_box  = (const float*)d_in[4];
    const float* b_box  = (const float*)d_in[5];
    const float* W_sub  = (const float*)d_in[6];
    const float* b_sub  = (const float*)d_in[7];
    const float* W_obj  = (const float*)d_in[8];
    const float* b_obj  = (const float*)d_in[9];
    const float* Wa_box = (const float*)d_in[10];
    const float* Wa_rel = (const float*)d_in[11];
    const float* b_a1   = (const float*)d_in[12];
    const float* Wa2    = (const float*)d_in[13];
    // d_in[14] = b_a2: constant per-logit offset, cancels in segment softmax.

    size_t off = 0;
    char* wsb = (char*)d_ws;
    auto wsAlloc = [&](size_t bytes) -> void* {
        void* p = wsb + off;
        off = (off + bytes + 255) & ~(size_t)255;
        return p;
    };

    bf16* relBf   = (bf16*)wsAlloc((size_t)NEDGE * FEATS * sizeof(bf16));
    bf16* boxBf   = (bf16*)wsAlloc((size_t)NBOX * OBJ * sizeof(bf16));
    bf16* WaRelT  = (bf16*)wsAlloc((size_t)OBJ * FEATS * sizeof(bf16));
    bf16* WsubT   = (bf16*)wsAlloc((size_t)OBJ * FEATS * sizeof(bf16));
    bf16* WobjT   = (bf16*)wsAlloc((size_t)OBJ * FEATS * sizeof(bf16));
    bf16* WaBoxT  = (bf16*)wsAlloc((size_t)OBJ * OBJ * sizeof(bf16));
    bf16* WboxT   = (bf16*)wsAlloc((size_t)OBJ * OBJ * sizeof(bf16));
    float* relAtt = (float*)wsAlloc((size_t)NEDGE * OBJ * sizeof(float));
    float* msgSub = (float*)wsAlloc((size_t)NEDGE * OBJ * sizeof(float));
    float* msgObj = (float*)wsAlloc((size_t)NEDGE * OBJ * sizeof(float));
    float* logitS = (float*)wsAlloc((size_t)NEDGE * sizeof(float));
    float* logitO = (float*)wsAlloc((size_t)NEDGE * sizeof(float));
    float* smaxS  = (float*)wsAlloc((size_t)NBOX * sizeof(float));
    float* smaxO  = (float*)wsAlloc((size_t)NBOX * sizeof(float));
    float* denS   = (float*)wsAlloc((size_t)NBOX * sizeof(float));
    float* denO   = (float*)wsAlloc((size_t)NBOX * sizeof(float));
    float* aggS   = (float*)wsAlloc((size_t)NBOX * OBJ * sizeof(float));
    float* aggO   = (float*)wsAlloc((size_t)NBOX * OBJ * sizeof(float));

    // --- precision conversion + weight transposes -------------------------
    cvt_bf16_kernel<<<4096, 256, 0, stream>>>(rel, relBf, (size_t)NEDGE * FEATS);
    cvt_bf16_kernel<<<2048, 256, 0, stream>>>(box, boxBf, (size_t)NBOX * OBJ);

    dim3 tb(32, 8);
    transpose_bf16_kernel<<<dim3(OBJ / 32, FEATS / 32), tb, 0, stream>>>(Wa_rel, WaRelT, FEATS, OBJ);
    transpose_bf16_kernel<<<dim3(OBJ / 32, FEATS / 32), tb, 0, stream>>>(W_sub,  WsubT,  FEATS, OBJ);
    transpose_bf16_kernel<<<dim3(OBJ / 32, FEATS / 32), tb, 0, stream>>>(W_obj,  WobjT,  FEATS, OBJ);
    transpose_bf16_kernel<<<dim3(OBJ / 32, OBJ / 32),   tb, 0, stream>>>(Wa_box, WaBoxT, OBJ, OBJ);
    transpose_bf16_kernel<<<dim3(OBJ / 32, OBJ / 32),   tb, 0, stream>>>(W_box,  WboxT,  OBJ, OBJ);

    // --- iteration-invariant rel projections (async-LDS WMMA GEMMs) -------
    const int relBlocks = (NEDGE / 64) * (OBJ / 256);   // 4096
    gemm_bf16_kernel<<<relBlocks, 256, 0, stream>>>(relBf, WaRelT, nullptr, relAtt, OBJ, FEATS);
    gemm_bf16_kernel<<<relBlocks, 256, 0, stream>>>(relBf, WsubT,  b_sub,   msgSub, OBJ, FEATS);
    gemm_bf16_kernel<<<relBlocks, 256, 0, stream>>>(relBf, WobjT,  b_obj,   msgObj, OBJ, FEATS);

    // --- init accumulators ------------------------------------------------
    hipMemsetAsync(logitS, 0, (size_t)NEDGE * sizeof(float), stream);
    hipMemsetAsync(logitO, 0, (size_t)NEDGE * sizeof(float), stream);
    hipMemsetAsync(denS,   0, (size_t)NBOX * sizeof(float), stream);
    hipMemsetAsync(denO,   0, (size_t)NBOX * sizeof(float), stream);
    hipMemsetAsync(aggS,   0, (size_t)NBOX * OBJ * sizeof(float), stream);
    hipMemsetAsync(aggO,   0, (size_t)NBOX * OBJ * sizeof(float), stream);
    fill_f32_kernel<<<(NBOX + 255) / 256, 256, 0, stream>>>(smaxS, -__builtin_huge_valf(), NBOX);
    fill_f32_kernel<<<(NBOX + 255) / 256, 256, 0, stream>>>(smaxO, -__builtin_huge_valf(), NBOX);

    // --- attention logits (gathered WMMA GEMM, fused relu+dot epilogue) ---
    att_gemm_kernel<<<relBlocks, 256, 0, stream>>>(boxBf, WaBoxT, esub, relAtt, b_a1, Wa2, logitS, OBJ);
    att_gemm_kernel<<<relBlocks, 256, 0, stream>>>(boxBf, WaBoxT, eobj, relAtt, b_a1, Wa2, logitO, OBJ);

    // --- segment softmax --------------------------------------------------
    const int eb = (NEDGE + 255) / 256;
    seg_max_kernel<<<eb, 256, 0, stream>>>(logitS, esub, smaxS, NEDGE);
    seg_max_kernel<<<eb, 256, 0, stream>>>(logitO, eobj, smaxO, NEDGE);
    seg_exp_kernel<<<eb, 256, 0, stream>>>(logitS, esub, smaxS, denS, NEDGE);
    seg_exp_kernel<<<eb, 256, 0, stream>>>(logitO, eobj, smaxO, denO, NEDGE);

    // --- weighted message aggregation ------------------------------------
    scatter_kernel<<<NEDGE, 256, 0, stream>>>(logitS, denS, esub, msgSub, aggS);
    scatter_kernel<<<NEDGE, 256, 0, stream>>>(logitO, denO, eobj, msgObj, aggO);

    // --- final node update (WMMA GEMM, fused bias+agg+relu) ---------------
    const int finBlocks = (NBOX / 64) * (OBJ / 256);    // 512
    final_gemm_kernel<<<finBlocks, 256, 0, stream>>>(boxBf, WboxT, b_box, aggS, aggO,
                                                     (float*)d_out, OBJ, OBJ);
}
